// LSTMModel_28174985462335
// MI455X (gfx1250) — matrix operand order repaired and checked by reading, not compiled, not run on hardware
//
#include <hip/hip_runtime.h>

#define Bsz   64
#define Tlen  256
#define INPUT 128
#define Hd    512
#define G4    2048
#define NL    6
#define NWG   16          // scan workgroups; each owns HCH h-columns
#define HCH   32          // h-columns per scan WG (NWG*HCH == Hd)
#define MTOT  (Bsz * Tlen)

typedef __attribute__((ext_vector_type(16))) __bf16 bfrag;
typedef __attribute__((ext_vector_type(8)))  float  v8f;
typedef int v4i __attribute__((vector_size(16)));

union FragU { bfrag f; unsigned u[8]; };

#if defined(__HIP_DEVICE_COMPILE__)
#  if __has_builtin(__builtin_amdgcn_global_load_async_to_lds_b128)
#    define USE_ASYNC_LDS 1
#  endif
#endif

__device__ __forceinline__ void wait_async0() {
#if __has_builtin(__builtin_amdgcn_s_wait_asynccnt)
  __builtin_amdgcn_s_wait_asynccnt(0);
#else
  asm volatile("s_wait_asynccnt 0x0" ::: "memory");
#endif
}

// A-matrix 16x32 bf16 fragment (ISA 7.12.2): lane m = row, K striped over VGPRs.
__device__ __forceinline__ bfrag load_a_frag(const __bf16* A, int lda, int m0, int k0, int lane) {
  const int half = lane >> 4;
  const int m    = lane & 15;
  FragU r;
  const __bf16* row = A + (size_t)(m0 + m) * lda + k0 + half * 8;
#pragma unroll
  for (int v = 0; v < 8; ++v) {
    const int k = ((v & 4) ? 16 : 0) + (v & 3) * 2;   // + half*8 folded into row
    r.u[v] = *(const unsigned*)(row + k);
  }
  return r.f;
}

// B-matrix 32x16 bf16 fragment: B(k,n) = W[n,k] (row-major W, ldw = K).
__device__ __forceinline__ bfrag load_b_frag(const __bf16* W, int ldw, int n0, int k0, int lane) {
  const int half = lane >> 4;
  const int n    = lane & 15;
  FragU r;
  const __bf16* row = W + (size_t)(n0 + n) * ldw + k0 + half * 8;
#pragma unroll
  for (int v = 0; v < 8; ++v) r.u[v] = *(const unsigned*)(row + ((v & 4) ? 16 : 0) + (v & 3) * 2);
  return r.f;
}

__device__ __forceinline__ float sigmoidf_(float x) { return 1.0f / (1.0f + __expf(-x)); }

__device__ __forceinline__ void grid_barrier(unsigned* bar, unsigned target) {
  __threadfence();
  __syncthreads();
  if (threadIdx.x == 0) {
    atomicAdd(bar, 1u);
    while (__hip_atomic_load(bar, __ATOMIC_RELAXED, __HIP_MEMORY_SCOPE_AGENT) < target)
      __builtin_amdgcn_s_sleep(1);
  }
  __syncthreads();
  __threadfence();
}

// ---------- small prep kernels ----------
__global__ void f32_to_bf16_kernel(const float* __restrict__ in, __bf16* __restrict__ out, int n) {
  int i = blockIdx.x * 256 + threadIdx.x;
  if (i < n) out[i] = (__bf16)in[i];
}

__global__ void bias_add_kernel(const float* __restrict__ a, const float* __restrict__ b,
                                float* __restrict__ out, int n) {
  int i = blockIdx.x * 256 + threadIdx.x;
  if (i < n) out[i] = a[i] + b[i];
}

// ---------- input projection: xp[M, 4H] = X[M,K] @ W[4H,K]^T + bias ----------
// One wave computes a 16(M) x 64(N) strip as 4 accumulators sharing the A-frag.
__global__ __launch_bounds__(256) void lstm_proj_kernel(
    const __bf16* __restrict__ X, const __bf16* __restrict__ W,
    const float* __restrict__ bias, float* __restrict__ xp, int K) {
  const int lane  = threadIdx.x & 31;
  const int waveG = (blockIdx.x * 8) + (threadIdx.x >> 5);
  const int mtile = waveG >> 5;         // M/16 = 1024 tiles, 32 N-groups of 64
  const int ng    = waveG & 31;
  const int m0    = mtile * 16;
  const int n0    = ng * 64;
  const int half  = lane >> 4;

  v8f acc[4];
#pragma unroll
  for (int s = 0; s < 4; ++s) {
    const float bv = bias[n0 + s * 16 + (lane & 15)];
#pragma unroll
    for (int v = 0; v < 8; ++v) acc[s][v] = bv;
  }
#pragma unroll 2
  for (int k0 = 0; k0 < K; k0 += 32) {
    const bfrag a = load_a_frag(X, K, m0, k0, lane);
#pragma unroll
    for (int s = 0; s < 4; ++s) {
      const bfrag b = load_b_frag(W, K, n0 + s * 16, k0, lane);
      acc[s] = __builtin_amdgcn_wmma_f32_16x16x32_bf16(false, a, false, b,
                                                       (short)0, acc[s], false, false);
    }
  }
#pragma unroll
  for (int s = 0; s < 4; ++s) {
    const int n = n0 + s * 16 + (lane & 15);
#pragma unroll
    for (int v = 0; v < 8; ++v) {
      const int m = m0 + (half ? 8 : 0) + v;
      xp[(size_t)m * G4 + n] = acc[s][v];
    }
  }
}

// ---------- persistent recurrent scan ----------
// NWG WGs; WG wg owns h-columns [wg*HCH, wg*HCH+HCH). Cell state c lives in registers.
// Each wave owns one (mtile, coltile) job -> 64 WMMAs per step on the critical path.
__global__ __launch_bounds__(256) void lstm_scan_kernel(
    const float* __restrict__ xp,     // [B*T, 4H]
    const __bf16* __restrict__ Whh,   // [4H, H] bf16
    __bf16* __restrict__ hseq,        // [B, T, H] bf16 out
    __bf16* __restrict__ hbuf,        // [2, B, H] bf16 ping-pong
    unsigned* __restrict__ bar) {
  __shared__ __bf16 hlds[Bsz * Hd];   // 64 KB: full h staged per step
  const int tid   = threadIdx.x;
  const int lane  = tid & 31;
  const int wave  = tid >> 5;
  const int wg    = blockIdx.x;
  const int hbase = wg * HCH;
  const int half  = lane >> 4;

  // zero my slice of hbuf[0] (h at t=0 is zero)
  for (int i = tid; i < Bsz * HCH; i += 256) {
    const int r = i / HCH, c = i % HCH;
    hbuf[(size_t)r * Hd + hbase + c] = (__bf16)0.0f;
  }
  grid_barrier(bar, 1u * NWG);

  // this wave's job
  const int mtile = wave >> 1;               // 0..3
  const int ct    = wave & 1;                // 0..1
  const int m0    = mtile * 16;
  const int ncol  = hbase + ct * 16;

  v8f cst;
#pragma unroll
  for (int v = 0; v < 8; ++v) cst[v] = 0.0f;

  for (int t = 0; t < Tlen; ++t) {
    const __bf16* hcur = hbuf + (size_t)(t & 1) * Bsz * Hd;
    __bf16*       hnxt = hbuf + (size_t)((t + 1) & 1) * Bsz * Hd;

    // stage current h into LDS
#if defined(USE_ASYNC_LDS)
    {
      // builtin wants: (int4 __device__* global_src, int4 AS(3)* lds_dst, imm, cpol)
      auto gsrc = (__attribute__((address_space(1))) v4i*)(v4i*)hcur;
      auto ldst = (__attribute__((address_space(3))) v4i*)(v4i*)hlds;
      for (int i = tid; i < (Bsz * Hd) / 8; i += 256)
        __builtin_amdgcn_global_load_async_to_lds_b128(gsrc + i, ldst + i, 0, 0);
      wait_async0();
    }
#else
    {
      const uint4* src = (const uint4*)hcur;
      uint4*       dst = (uint4*)hlds;
      for (int i = tid; i < (Bsz * Hd) / 8; i += 256) dst[i] = src[i];
    }
#endif
    // prefetch next step's accumulator seeds while we compute this step
    if (t + 1 < Tlen) {
#pragma unroll
      for (int g = 0; g < 4; ++g)
        __builtin_prefetch(&xp[((size_t)m0 * Tlen + (t + 1)) * G4 + g * Hd + ncol], 0, 1);
    }
    __syncthreads();

    // accumulators seeded from xp (input proj + biases), gate order i,f,g,o
    v8f acc[4];
#pragma unroll
    for (int g = 0; g < 4; ++g) {
      const int n = g * Hd + ncol + (lane & 15);
#pragma unroll
      for (int v = 0; v < 8; ++v) {
        const int b = m0 + (half ? 8 : 0) + v;
        acc[g][v] = xp[((size_t)b * Tlen + t) * G4 + n];
      }
    }
#pragma unroll 2
    for (int k0 = 0; k0 < Hd; k0 += 32) {
      const bfrag a = load_a_frag(hlds, Hd, m0, k0, lane);
#pragma unroll
      for (int g = 0; g < 4; ++g) {
        const bfrag b = load_b_frag(Whh, Hd, g * Hd + ncol, k0, lane);
        acc[g] = __builtin_amdgcn_wmma_f32_16x16x32_bf16(false, a, false, b,
                                                         (short)0, acc[g], false, false);
      }
    }
    // gate nonlinearities + cell/hidden update (f32)
    const int col = ncol + (lane & 15);
#pragma unroll
    for (int v = 0; v < 8; ++v) {
      const float ig = sigmoidf_(acc[0][v]);
      const float fg = sigmoidf_(acc[1][v]);
      const float gg = tanhf(acc[2][v]);
      const float og = sigmoidf_(acc[3][v]);
      const float cc = fg * cst[v] + ig * gg;
      cst[v] = cc;
      const float hh = og * tanhf(cc);
      const int b = m0 + (half ? 8 : 0) + v;
      const __bf16 hb = (__bf16)hh;
      hnxt[(size_t)b * Hd + col] = hb;
      hseq[((size_t)b * Tlen + t) * Hd + col] = hb;
    }
    __syncthreads();                 // LDS reads done before next-step overwrite
    grid_barrier(bar, (unsigned)(t + 2) * NWG);
  }
}

// ---------- final FC on last timestep ----------
__global__ void lstm_fc_kernel(const __bf16* __restrict__ hseq, const float* __restrict__ fcw,
                               const float* __restrict__ fcb, float* __restrict__ out) {
  const int b = threadIdx.x;   // 0..63
  const int j = blockIdx.x;    // 0..9
  const __bf16* h = hseq + ((size_t)b * Tlen + (Tlen - 1)) * Hd;
  const float*  w = fcw + (size_t)j * Hd;
  float s = fcb[j];
  for (int k = 0; k < Hd; ++k) s += (float)h[k] * w[k];
  out[b * 10 + j] = s;
}

extern "C" void kernel_launch(void* const* d_in, const int* in_sizes, int n_in,
                              void* d_out, int out_size, void* d_ws, size_t ws_size,
                              hipStream_t stream) {
  (void)in_sizes; (void)n_in; (void)out_size; (void)ws_size;
  const float* x    = (const float*)d_in[0];
  const float* Wih0 = (const float*)d_in[1];
  const float* WihR = (const float*)d_in[2];
  const float* Whh  = (const float*)d_in[3];
  const float* bih  = (const float*)d_in[4];
  const float* bhh  = (const float*)d_in[5];
  const float* fcw  = (const float*)d_in[6];
  const float* fcb  = (const float*)d_in[7];

  char* p = (char*)d_ws;
  auto alloc = [&](size_t bytes) { char* r = p; p += (bytes + 255) & ~(size_t)255; return r; };
  __bf16* whh_bf  = (__bf16*)alloc((size_t)NL * G4 * Hd * 2);
  __bf16* wih0_bf = (__bf16*)alloc((size_t)G4 * INPUT * 2);
  __bf16* wihR_bf = (__bf16*)alloc((size_t)(NL - 1) * G4 * Hd * 2);
  float*  bias    = (float*)alloc((size_t)NL * G4 * 4);
  __bf16* x0_bf   = (__bf16*)alloc((size_t)MTOT * INPUT * 2);
  __bf16* hseqA   = (__bf16*)alloc((size_t)MTOT * Hd * 2);
  __bf16* hseqB   = (__bf16*)alloc((size_t)MTOT * Hd * 2);
  float*  xp      = (float*)alloc((size_t)MTOT * G4 * 4);
  __bf16* hbuf    = (__bf16*)alloc((size_t)2 * Bsz * Hd * 2);
  unsigned* bar   = (unsigned*)alloc(256);

  auto cvt = [&](const float* in, __bf16* out, int n) {
    f32_to_bf16_kernel<<<(n + 255) / 256, 256, 0, stream>>>(in, out, n);
  };
  cvt(Whh,  whh_bf,  NL * G4 * Hd);
  cvt(Wih0, wih0_bf, G4 * INPUT);
  cvt(WihR, wihR_bf, (NL - 1) * G4 * Hd);
  cvt(x,    x0_bf,   MTOT * INPUT);
  bias_add_kernel<<<(NL * G4 + 255) / 256, 256, 0, stream>>>(bih, bhh, bias, NL * G4);

  for (int l = 0; l < NL; ++l) {
    const __bf16* Xin  = (l == 0) ? x0_bf : ((l & 1) ? hseqA : hseqB);
    __bf16*       Hout = (l & 1) ? hseqB : hseqA;
    const int K        = (l == 0) ? INPUT : Hd;
    const __bf16* Wih  = (l == 0) ? wih0_bf : (wihR_bf + (size_t)(l - 1) * G4 * Hd);
    const int blocks   = (MTOT / 16) * (G4 / 64) / 8;   // 4096
    lstm_proj_kernel<<<blocks, 256, 0, stream>>>(Xin, Wih, bias + (size_t)l * G4, xp, K);
    (void)hipMemsetAsync(bar, 0, 16, stream);
    lstm_scan_kernel<<<NWG, 256, 0, stream>>>(xp, whh_bf + (size_t)l * G4 * Hd, Hout, hbuf, bar);
  }
  lstm_fc_kernel<<<10, 64, 0, stream>>>(hseqB, fcw, fcb, (float*)d_out);
}
